// SelfAttention_7249904796361
// MI455X (gfx1250) — compile-verified, matrix operand order checked by reading, not run on hardware
//
#include <hip/hip_runtime.h>
#include <hip/hip_bf16.h>

// ---------------------------------------------------------------------------
// Self-attention forward for B=2, S=2048, D=1024, H=16, Hd=64 (fp32 ref).
// Phase 1: QKV projections (f32 -> f16 GEMM w/ f32 WMMA accum) into workspace.
// Phase 2: flash-style attention, v_wmma_f32_16x16x32_f16 for QK^T and PV,
//          K tiles staged by the Tensor Data Mover (hardware-padded rows),
//          double-buffered so DMA/loads overlap the matrix math.
// ---------------------------------------------------------------------------

typedef __attribute__((ext_vector_type(16))) _Float16 v16h;
typedef __attribute__((ext_vector_type(8)))  _Float16 v8h;
typedef __attribute__((ext_vector_type(8)))  float    v8f;
typedef __attribute__((ext_vector_type(4)))  unsigned int v4u;
typedef __attribute__((ext_vector_type(8)))  int      v8i;
typedef __attribute__((ext_vector_type(4)))  int      v4i;

#define NB  2
#define NS  2048
#define ND  1024
#define NH  16
#define HD  64
#define BH  (NB * NH)          // 32
#define BS  (NB * NS)          // 4096

// ---- WMMA helper -----------------------------------------------------------
__device__ __forceinline__ v8f wmma_f16(v16h a, v16h b, v8f c) {
  // D = A(16x32 f16) * B(32x16 f16) + C(16x16 f32)
  return __builtin_amdgcn_wmma_f32_16x16x32_f16(
      /*neg_a=*/false, a, /*neg_b=*/false, b,
      /*c_mod=*/(short)0, c, /*reuse_a=*/false, /*reuse_b=*/false);
}

// Fragment load for memory laid out contiguous along K per row:
//   A-fragment: "row" = matrix row m        (memory [m][k])
//   B-fragment: "row" = matrix column n     (memory [n][k], i.e. transposed)
// lane holds row = row_off + (lane&15); halves j -> K = j + (j&8) + 8*(lane>=16)
// = two contiguous 8-half (16B) chunks at half-offsets lh8 and 16+lh8.
__device__ __forceinline__ v16h frag_load_16x32(const _Float16* base,
                                                int stride_h, int row_off) {
  const int lane = threadIdx.x & 31;
  const int lh8  = (lane & 16) >> 1;                 // 0 or 8
  const _Float16* p = base + (size_t)(row_off + (lane & 15)) * stride_h;
  const v8h c0 = *(const v8h*)(p + lh8);
  const v8h c1 = *(const v8h*)(p + 16 + lh8);
  return __builtin_shufflevector(c0, c1, 0, 1, 2, 3, 4, 5, 6, 7,
                                 8, 9, 10, 11, 12, 13, 14, 15);
}

// ---------------------------------------------------------------------------
// Kernel 1: y = x @ W + b, stored f16 in [bh][s][d] layout.
// grid = (N/128, M/128, 3), block = 256 (8 waves). Wave tile 32x64 (8 WMMAs).
// W tile stored transposed ([n][k]) in LDS so B fragments load contiguously.
// ---------------------------------------------------------------------------
__global__ __launch_bounds__(256)
void qkv_gemm_kernel(const float* __restrict__ x,
                     const float* __restrict__ Wq, const float* __restrict__ bq,
                     const float* __restrict__ Wk, const float* __restrict__ bk,
                     const float* __restrict__ Wv, const float* __restrict__ bv,
                     _Float16* __restrict__ qo, _Float16* __restrict__ ko,
                     _Float16* __restrict__ vo) {
  const int which = blockIdx.z;
  const float* W    = (which == 0) ? Wq : (which == 1) ? Wk : Wv;
  const float* bias = (which == 0) ? bq : (which == 1) ? bk : bv;
  _Float16*    dst  = (which == 0) ? qo : (which == 1) ? ko : vo;

  const int mbase = blockIdx.y * 128;
  const int nbase = blockIdx.x * 128;
  const int tid   = threadIdx.x;
  const int lane  = tid & 31;
  const int w     = tid >> 5;
  const int wm    = (w >> 1) * 32;   // 0,32,64,96
  const int wn    = (w & 1) * 64;    // 0,64

  __shared__ alignas(16) _Float16 xs[128][40];    // x tile   [m][k] (+pad)
  __shared__ alignas(16) _Float16 wt[128][40];    // W tile^T [n][k] (+pad)

  v8f acc[2][4];
#pragma unroll
  for (int i = 0; i < 2; ++i)
#pragma unroll
    for (int c = 0; c < 4; ++c) acc[i][c] = (v8f){0, 0, 0, 0, 0, 0, 0, 0};

  for (int kb = 0; kb < ND; kb += 32) {
    __syncthreads();
    {  // x tile: 128 rows x 8 float4 -> 4 per thread (coalesced)
      int i = tid;
#pragma unroll
      for (int it = 0; it < 4; ++it, i += 256) {
        const int r = i >> 3, c4 = (i & 7) * 4;
        const float4 vv = *(const float4*)(x + (size_t)(mbase + r) * ND + kb + c4);
        xs[r][c4 + 0] = (_Float16)vv.x; xs[r][c4 + 1] = (_Float16)vv.y;
        xs[r][c4 + 2] = (_Float16)vv.z; xs[r][c4 + 3] = (_Float16)vv.w;
      }
    }
    {  // W tile: coalesced global read, transposed scatter into wt[n][k]
      int i = tid;
#pragma unroll
      for (int it = 0; it < 4; ++it, i += 256) {
        const int r = i >> 5, c4 = (i & 31) * 4;
        const float4 vv = *(const float4*)(W + (size_t)(kb + r) * ND + nbase + c4);
        wt[c4 + 0][r] = (_Float16)vv.x; wt[c4 + 1][r] = (_Float16)vv.y;
        wt[c4 + 2][r] = (_Float16)vv.z; wt[c4 + 3][r] = (_Float16)vv.w;
      }
    }
    __syncthreads();

    const v16h a0 = frag_load_16x32(&xs[wm][0], 40, 0);
    const v16h a1 = frag_load_16x32(&xs[wm][0], 40, 16);
#pragma unroll
    for (int cc = 0; cc < 4; ++cc) {
      const v16h b = frag_load_16x32(&wt[wn + cc * 16][0], 40, 0);
      acc[0][cc] = wmma_f16(a0, b, acc[0][cc]);
      acc[1][cc] = wmma_f16(a1, b, acc[1][cc]);
    }
  }

  // Epilogue: bias add, f16 store into [bh][s][d] layout.
  const int lh8 = (lane & 16) >> 1;
#pragma unroll
  for (int i = 0; i < 2; ++i) {
#pragma unroll
    for (int cc = 0; cc < 4; ++cc) {
      const int n  = nbase + wn + cc * 16 + (lane & 15);
      const float bb = bias[n];
      const int h = n >> 6, d = n & 63;
#pragma unroll
      for (int j = 0; j < 8; ++j) {
        const int m = mbase + wm + i * 16 + j + lh8;
        const int b = m >> 11, s = m & (NS - 1);
        dst[(((size_t)(b * NH + h) * NS + s) << 6) + d] =
            (_Float16)(acc[i][cc][j] + bb);
      }
    }
  }
}

// ---------------------------------------------------------------------------
// Kernel 2: flash attention. grid = (S/128, BH), block = 256 (8 waves).
// Each wave: 16 queries x Hd=64; key loop in tiles of 32, online softmax.
// K tiles DMA'd by the TDM with hardware row padding; V tiles transposed
// through VGPRs; both double-buffered (one barrier per tile).
// ---------------------------------------------------------------------------
__global__ __launch_bounds__(256)
void attn_kernel(const _Float16* __restrict__ qf,
                 const _Float16* __restrict__ kf,
                 const _Float16* __restrict__ vf,
                 float* __restrict__ out) {
  const int bh    = blockIdx.y;
  const int qbase = blockIdx.x * 128;
  const int tid   = threadIdx.x;
  const int lane  = tid & 31;
  const int w     = tid >> 5;
  const int lh8   = (lane & 16) >> 1;

  __shared__ alignas(16) _Float16 kt[2][32][72];    // K tile   [key][d] (+pad)
  __shared__ alignas(16) _Float16 vtT[2][64][40];   // V tile^T [d][key] (+pad)
  __shared__ alignas(16) _Float16 ps[8][16][40];    // per-wave P staging

  const _Float16* qbh = qf + (size_t)bh * NS * HD;
  const _Float16* kbh = kf + (size_t)bh * NS * HD;
  const _Float16* vbh = vf + (size_t)bh * NS * HD;

  // ---- staging helpers ----------------------------------------------------
  // TDM: DMA a 32x64-half K tile into kt[buf], padding each 128B row by 16B
  // so the LDS row stride is 144B (= kt's 72-half stride). Wave 0 issues.
  auto stage_k = [&](int buf, int kb) {
#if __has_builtin(__builtin_amdgcn_tensor_load_to_lds)
    if (w == 0) {
      const unsigned long long ga =
          (unsigned long long)(uintptr_t)(kbh + (size_t)kb * HD);
      const unsigned int lds = (unsigned int)(uintptr_t)&kt[buf][0][0];
      v4u g0;
      g0[0] = 1u;                                    // count=1 (valid D#)
      g0[1] = lds;                                   // lds_addr
      g0[2] = (unsigned int)ga;                      // global_addr[31:0]
      g0[3] = ((unsigned int)(ga >> 32) & 0x01FFFFFFu) | 0x80000000u; // +type=2
      v8i g1;
      g1[0] = 0x07110000;        // data_size=2B, pad_enable, intvl=32dw, pad=4dw
      g1[1] = (int)(64u << 16);  // tensor_dim0 = 64 elements/row
      g1[2] = (int)(32u << 16);  // tensor_dim1 = 32 rows
      g1[3] = (int)(64u << 16);  // tile_dim0 = 64
      g1[4] = 32;                // tile_dim1 = 32 (tile_dim2 = 0)
      g1[5] = 64;                // tensor_dim0_stride = 64 elements
      g1[6] = 0;
      g1[7] = 0;
      const v4i gz = {0, 0, 0, 0};
#if __clang_major__ >= 23
      const v8i gz8 = {0, 0, 0, 0, 0, 0, 0, 0};
      __builtin_amdgcn_tensor_load_to_lds(g0, g1, gz, gz, gz8, 0);
#else
      __builtin_amdgcn_tensor_load_to_lds(g0, g1, gz, gz, 0);
#endif
    }
#else
    // Fallback: cooperative synchronous copy (256 threads x 16B).
    const int r = tid >> 3, c8 = (tid & 7) * 8;
    *(v8h*)&kt[buf][r][c8] = *(const v8h*)(kbh + (size_t)(kb + r) * HD + c8);
#endif
  };
  auto tdm_wait = [&]() {
#if __has_builtin(__builtin_amdgcn_tensor_load_to_lds)
    if (w == 0) __builtin_amdgcn_s_wait_tensorcnt(0);
#endif
  };
  auto v_fetch = [&](int kb) -> v8h {   // coalesced row fetch of V tile
    return *(const v8h*)(vbh + (size_t)(kb + (tid >> 3)) * HD + (tid & 7) * 8);
  };
  auto v_scatter = [&](int buf, v8h vv) {  // transpose into vtT[buf][d][key]
    const int r = tid >> 3, c8 = (tid & 7) * 8;
#pragma unroll
    for (int ii = 0; ii < 8; ++ii) vtT[buf][c8 + ii][r] = vv[ii];
  };

  // ---- per-wave state -----------------------------------------------------
  const int qrow0 = qbase + w * 16;
  const v16h aq0 = frag_load_16x32(qbh + (size_t)qrow0 * HD, HD, 0);
  const v16h aq1 = frag_load_16x32(qbh + (size_t)qrow0 * HD + 32, HD, 0);

  v8f acc[4];
#pragma unroll
  for (int t = 0; t < 4; ++t) acc[t] = (v8f){0, 0, 0, 0, 0, 0, 0, 0};
  float mrun[8], lrun[8];
#pragma unroll
  for (int j = 0; j < 8; ++j) { mrun[j] = -3.0e38f; lrun[j] = 0.0f; }

  // Prologue: stage tile 0 into buffer 0.
  stage_k(0, 0);
  v_scatter(0, v_fetch(0));
  tdm_wait();
  __syncthreads();

  int cur = 0;
  for (int kb = 0; kb < NS; kb += 32) {
    const int nxt = cur ^ 1;
    const bool have_next = (kb + 32) < NS;
    v8h vnext = (v8h){0, 0, 0, 0, 0, 0, 0, 0};
    if (have_next) {          // kick off next tile before computing this one
      stage_k(nxt, kb + 32);
      vnext = v_fetch(kb + 32);
    }

    // ---- compute on buffer `cur` ----
    const v16h bk00 = frag_load_16x32(&kt[cur][0][0],  72, 0);   // keys 0-15
    const v16h bk01 = frag_load_16x32(&kt[cur][0][32], 72, 0);
    const v16h bk10 = frag_load_16x32(&kt[cur][0][0],  72, 16);  // keys 16-31
    const v16h bk11 = frag_load_16x32(&kt[cur][0][32], 72, 16);
    const v8f zero = (v8f){0, 0, 0, 0, 0, 0, 0, 0};
    v8f s0 = wmma_f16(aq0, bk00, zero); s0 = wmma_f16(aq1, bk01, s0);
    v8f s1 = wmma_f16(aq0, bk10, zero); s1 = wmma_f16(aq1, bk11, s1);

    // Online softmax (row groups: lanes 0-15 rows j, lanes 16-31 rows j+8).
    float scale[8];
#pragma unroll
    for (int j = 0; j < 8; ++j) {
      const float t0 = s0[j], t1 = s1[j];
      float mx = fmaxf(t0, t1);
      mx = fmaxf(mx, __shfl_xor(mx, 1, 32));
      mx = fmaxf(mx, __shfl_xor(mx, 2, 32));
      mx = fmaxf(mx, __shfl_xor(mx, 4, 32));
      mx = fmaxf(mx, __shfl_xor(mx, 8, 32));
      const float mnew = fmaxf(mrun[j], mx);
      const float sc = __expf(mrun[j] - mnew);
      const float p0 = __expf(t0 - mnew);
      const float p1 = __expf(t1 - mnew);
      float rs = p0 + p1;
      rs += __shfl_xor(rs, 1, 32);
      rs += __shfl_xor(rs, 2, 32);
      rs += __shfl_xor(rs, 4, 32);
      rs += __shfl_xor(rs, 8, 32);
      lrun[j] = lrun[j] * sc + rs;
      mrun[j] = mnew;
      scale[j] = sc;
      const int row = j + lh8;               // C-layout row this lane holds
      ps[w][row][(lane & 15)]      = (_Float16)p0;   // keys kb+0..15
      ps[w][row][16 + (lane & 15)] = (_Float16)p1;   // keys kb+16..31
    }
#pragma unroll
    for (int t = 0; t < 4; ++t)
#pragma unroll
      for (int j = 0; j < 8; ++j) acc[t][j] *= scale[j];

    // O += P(16x32) * V(32x64); P round-trips LDS (same wave), V from ^T tile.
    const v16h pa = frag_load_16x32(&ps[w][0][0], 40, 0);
#pragma unroll
    for (int t = 0; t < 4; ++t) {
      const v16h bv_ = frag_load_16x32(&vtT[cur][t * 16][0], 40, 0);
      acc[t] = wmma_f16(pa, bv_, acc[t]);
    }

    // ---- finish staging of next tile, single barrier per iteration ----
    if (have_next) v_scatter(nxt, vnext);
    tdm_wait();
    __syncthreads();
    cur = nxt;
  }

  // Epilogue: normalize and scatter to [B, S, H*Hd] fp32 output.
  const int b = bh >> 4, h = bh & 15;
#pragma unroll
  for (int t = 0; t < 4; ++t) {
    const int d = t * 16 + (lane & 15);
#pragma unroll
    for (int j = 0; j < 8; ++j) {
      const int srow = qrow0 + j + lh8;
      out[(size_t)(b * NS + srow) * ND + h * HD + d] = acc[t][j] / lrun[j];
    }
  }
}

// ---------------------------------------------------------------------------
extern "C" void kernel_launch(void* const* d_in, const int* in_sizes, int n_in,
                              void* d_out, int out_size, void* d_ws, size_t ws_size,
                              hipStream_t stream) {
  (void)in_sizes; (void)n_in; (void)out_size; (void)ws_size;
  const float* x  = (const float*)d_in[0];
  const float* Wq = (const float*)d_in[1];
  const float* bq = (const float*)d_in[2];
  const float* Wk = (const float*)d_in[3];
  const float* bk = (const float*)d_in[4];
  const float* Wv = (const float*)d_in[5];
  const float* bv = (const float*)d_in[6];
  float* out = (float*)d_out;

  const size_t perMat = (size_t)BH * NS * HD;   // 4,194,304 halves (8 MB)
  _Float16* q = (_Float16*)d_ws;
  _Float16* k = q + perMat;
  _Float16* v = k + perMat;

  // Phase 1: Q/K/V projections (z selects which weight matrix).
  qkv_gemm_kernel<<<dim3(ND / 128, BS / 128, 3), 256, 0, stream>>>(
      x, Wq, bq, Wk, bk, Wv, bv, q, k, v);

  // Phase 2: flash attention per (bh, 128-query block).
  attn_kernel<<<dim3(NS / 128, BH), 256, 0, stream>>>(q, k, v, out);
}
